// EvoformerBlock_66606352827379
// MI455X (gfx1250) — compile-verified, matrix operand order checked by reading, not run on hardware
//
#include <hip/hip_runtime.h>
#include <hip/hip_bf16.h>
#include <math.h>

// =============================================================================
// Evoformer block for gfx1250 (MI455X).  All contractions run on
// v_wmma_f32_16x16x32_f16 (f16 inputs, f32 accumulate).  Attention and the
// outer-product-mean are fused so the big [S,H,N,N] logits (256MB) and the
// [N*32, N*32] outer tensor (268MB) are never materialized.
// GEMM is double-buffered (2x12KB LDS) with register prefetch so global_load
// traffic overlaps the WMMAs; fragment reads are 2x ds_load_b128 per lane;
// the N-edge guard is a block-uniform scalar branch (no exec-mask loops in
// the hot path).  Workspace requirement: ~355 MB.
// =============================================================================

typedef __attribute__((ext_vector_type(16))) _Float16 v16h;
typedef __attribute__((ext_vector_type(8)))  _Float16 v8h;
typedef __attribute__((ext_vector_type(4)))  _Float16 v4h;
typedef __attribute__((ext_vector_type(8)))  float    v8f;
typedef __attribute__((ext_vector_type(4)))  float    v4f;
typedef __attribute__((ext_vector_type(16))) float    v16f;

static __device__ __forceinline__ v8f wmma16(v16h a, v16h b, v8f c) {
  return __builtin_amdgcn_wmma_f32_16x16x32_f16(false, a, false, b, (short)0, c,
                                                false, false);
}
static __device__ __forceinline__ float sigmoidf_(float x) {
  return 1.0f / (1.0f + __expf(-x));
}

// 16 contiguous halves (32B, 16B-aligned) -> v16h   (2x b128 load)
static __device__ __forceinline__ v16h ld16h(const _Float16* p) {
  union { v16h v; v8h h[2]; } u;
  const v8h* q = (const v8h*)p;
  u.h[0] = q[0];
  u.h[1] = q[1];
  return u.v;
}
// 16 contiguous floats (64B, 16B-aligned) -> v16h   (4x b128 load + packed cvt)
static __device__ __forceinline__ v16h ld16f(const float* p) {
  union { v16f v; v4f q[4]; } u;
  const v4f* g = (const v4f*)p;
  u.q[0] = g[0]; u.q[1] = g[1]; u.q[2] = g[2]; u.q[3] = g[3];
  return __builtin_convertvector(u.v, v16h);
}

// ---------------- LayerNorm: one wave32 per row ------------------------------
__global__ __launch_bounds__(256) void ln_kernel(const float* __restrict__ x,
                                                 const float* __restrict__ w,
                                                 const float* __restrict__ b,
                                                 float* __restrict__ out,
                                                 int R, int C) {
  int wv = threadIdx.x >> 5, lane = threadIdx.x & 31;
  int row = blockIdx.x * 8 + wv;
  if (row >= R) return;
  const float* xr = x + (size_t)row * C;
  float s = 0.f;
  for (int c = lane; c < C; c += 32) s += xr[c];
#pragma unroll
  for (int o = 16; o; o >>= 1) s += __shfl_xor(s, o, 32);
  float mu = s / C;
  float v = 0.f;
  for (int c = lane; c < C; c += 32) { float d = xr[c] - mu; v += d * d; }
#pragma unroll
  for (int o = 16; o; o >>= 1) v += __shfl_xor(v, o, 32);
  float inv = rsqrtf(v / C + 1e-5f);
  float* orow = out + (size_t)row * C;
  for (int c = lane; c < C; c += 32) orow[c] = (xr[c] - mu) * inv * w[c] + b[c];
}

// ---------------- Generic WMMA GEMM with fused epilogue ----------------------
// out[M,N] = epilogue( A[M,K] @ W[K,N] )    REQUIRES: M%64==0, K%32==0, N%4==0
// epilogue: (+bias[col]) * scale -> optional relu -> * rowmask[row] -> +residual
// Double-buffered LDS + register prefetch: one barrier per k-step; global
// loads for tile k+1 are issued before the WMMAs of tile k.
__global__ __launch_bounds__(256) void gemm_kernel(const float* __restrict__ A,
                                                   const float* __restrict__ W,
                                                   const float* __restrict__ bias,
                                                   const float* __restrict__ rowmask,
                                                   const float* __restrict__ residual,
                                                   float* __restrict__ out,
                                                   int M, int N, int K,
                                                   float scale, int relu) {
  __shared__ _Float16 As[2][64][48];   // [buf][row][k]  row stride 96B
  __shared__ _Float16 BsT[2][64][48];  // [buf][col][k]  (B tile transposed)
  int t = threadIdx.x;
  int lane = t & 31, wv = t >> 5;
  int wm = wv & 3, wn = wv >> 2;       // 4x2 wave grid over 64x64 tile
  int fr = lane & 15, fk = (lane >> 4) << 4;
  int row0 = blockIdx.y * 64, col0 = blockIdx.x * 64;
  // per-thread tile-load coordinates (A: 2 float4 groups, B: 2 float4 groups)
  int ar = t >> 3, ac = (t & 7) << 2;      // A rows ar, ar+32; cols ac..ac+3
  int br = t >> 4, bc = (t & 15) << 2;     // B rows br, br+16; cols bc..bc+3
  bool fullN = (col0 + 64 <= N);           // block-uniform: no exec masking
  v8f acc0 = {}, acc1 = {};
  int kb = K >> 5;
  v4f aR0, aR1, bR0, bR1;

  auto loadTiles = [&](int k0) {
    aR0 = *(const v4f*)&A[(size_t)(row0 + ar) * K + k0 + ac];
    aR1 = *(const v4f*)&A[(size_t)(row0 + ar + 32) * K + k0 + ac];
    if (fullN) {
      bR0 = *(const v4f*)&W[(size_t)(k0 + br) * N + col0 + bc];
      bR1 = *(const v4f*)&W[(size_t)(k0 + br + 16) * N + col0 + bc];
    } else {
#pragma unroll
      for (int j = 0; j < 4; ++j) {
        int gc = col0 + bc + j;
        bR0[j] = (gc < N) ? W[(size_t)(k0 + br) * N + gc] : 0.f;
        bR1[j] = (gc < N) ? W[(size_t)(k0 + br + 16) * N + gc] : 0.f;
      }
    }
  };
  auto storeTiles = [&](int buf) {
    *(v4h*)&As[buf][ar][ac]      = __builtin_convertvector(aR0, v4h);
    *(v4h*)&As[buf][ar + 32][ac] = __builtin_convertvector(aR1, v4h);
    v4h h0 = __builtin_convertvector(bR0, v4h);
    v4h h1 = __builtin_convertvector(bR1, v4h);
#pragma unroll
    for (int j = 0; j < 4; ++j) {
      BsT[buf][bc + j][br]      = h0[j];
      BsT[buf][bc + j][br + 16] = h1[j];
    }
  };

  loadTiles(0);
  storeTiles(0);
  __syncthreads();
  for (int k = 0; k < kb; ++k) {
    int buf = k & 1;
    if (k + 1 < kb) loadTiles((k + 1) << 5);           // prefetch next tile
    v16h af = ld16h(&As[buf][wm * 16 + fr][fk]);
    v16h b0 = ld16h(&BsT[buf][wn * 32 + fr][fk]);
    v16h b1 = ld16h(&BsT[buf][wn * 32 + 16 + fr][fk]);
    acc0 = wmma16(af, b0, acc0);
    acc1 = wmma16(af, b1, acc1);
    if (k + 1 < kb) storeTiles(buf ^ 1);
    __syncthreads();
  }
  int gc0 = col0 + wn * 32 + fr;
  int gc1 = gc0 + 16;
#pragma unroll
  for (int r = 0; r < 8; ++r) {
    int gr = row0 + wm * 16 + r + ((lane >> 4) << 3);
    float rm = rowmask ? rowmask[gr] : 1.0f;
    if (gc0 < N) {
      float v = acc0[r];
      if (bias) v += bias[gc0];
      v *= scale;
      if (relu) v = fmaxf(v, 0.f);
      v *= rm;
      size_t o = (size_t)gr * N + gc0;
      if (residual) v += residual[o];
      out[o] = v;
    }
    if (gc1 < N) {
      float v = acc1[r];
      if (bias) v += bias[gc1];
      v *= scale;
      if (relu) v = fmaxf(v, 0.f);
      v *= rm;
      size_t o = (size_t)gr * N + gc1;
      if (residual) v += residual[o];
      out[o] = v;
    }
  }
}

// ---------------- Fused gated-attention core ---------------------------------
// q,k,v layout [Bdim*len, H*32].  One wave per (b, h, 16-query tile).
// Phase1: logits = QK^T (+bias1[b,k] +bias2[(q*K+k)*H+h]) -> LDS
// Phase2: row softmax (lanes 0..15, one row each)
// Phase3: O = P V
__global__ __launch_bounds__(32) void attn_kernel(const float* __restrict__ q,
                                                  const float* __restrict__ k,
                                                  const float* __restrict__ v,
                                                  const float* __restrict__ bias1,
                                                  const float* __restrict__ bias2,
                                                  float* __restrict__ out,
                                                  int Bdim, int Qlen, int Klen, int H) {
  __shared__ float P[16][260];       // row stride 1040B (16B aligned)
  int b = blockIdx.z, h = blockIdx.y, q0 = blockIdx.x * 16;
  int lane = threadIdx.x;
  int fr = lane & 15, fk = (lane >> 4) << 4;
  int HC = H * 32;
  v16h qf = ld16f(&q[((size_t)(b * Qlen + q0 + fr) * HC) + h * 32 + fk]);
  for (int kt = 0; kt < (Klen >> 4); ++kt) {
    v16h kf = ld16f(&k[((size_t)(b * Klen + kt * 16 + fr) * HC) + h * 32 + fk]);
    v8f acc = {};
    acc = wmma16(qf, kf, acc);
    int key = kt * 16 + fr;
    float b1v = bias1 ? bias1[(size_t)b * Klen + key] : 0.f;
#pragma unroll
    for (int r = 0; r < 8; ++r) {
      int row = r + ((lane >> 4) << 3);
      float bv = acc[r] + b1v;
      if (bias2) bv += bias2[((size_t)(q0 + row) * Klen + key) * H + h];
      P[row][key] = bv;
    }
  }
  __syncthreads();
  if (lane < 16) {
    float mx = -3.0e38f;
    for (int j = 0; j < Klen; ++j) mx = fmaxf(mx, P[lane][j]);
    float s = 0.f;
    for (int j = 0; j < Klen; ++j) { float e = __expf(P[lane][j] - mx); P[lane][j] = e; s += e; }
    float inv = 1.0f / s;
    for (int j = 0; j < Klen; ++j) P[lane][j] *= inv;
  }
  __syncthreads();
  v8f o0 = {}, o1 = {};
  for (int kb = 0; kb < (Klen >> 5); ++kb) {
    v16h pf = ld16f(&P[fr][kb * 32 + fk]);
    v16h v0, v1;
#pragma unroll
    for (int i = 0; i < 16; ++i) {
      size_t vb = ((size_t)(b * Klen + kb * 32 + fk + i) * HC) + h * 32;
      v0[i] = (_Float16)v[vb + fr];
      v1[i] = (_Float16)v[vb + 16 + fr];
    }
    o0 = wmma16(pf, v0, o0);
    o1 = wmma16(pf, v1, o1);
  }
#pragma unroll
  for (int r = 0; r < 8; ++r) {
    int row = r + ((lane >> 4) << 3);
    size_t ob = ((size_t)(b * Qlen + q0 + row) * HC) + h * 32;
    out[ob + fr]      = o0[r];
    out[ob + 16 + fr] = o1[r];
  }
}

// ---------------- Triangle-mul einsum: per-channel WMMA GEMM -----------------
// outgoing: x[i,j,c] = sum_k a[i,k,c]*b[j,k,c]
// incoming: x[i,j,c] = sum_k a[k,i,c]*b[k,j,c]
__global__ __launch_bounds__(32) void trimul_kernel(const float* __restrict__ a,
                                                    const float* __restrict__ b,
                                                    float* __restrict__ x,
                                                    int N, int C, int outgoing) {
  int c = blockIdx.z;
  int i0 = blockIdx.x * 16, j0 = blockIdx.y * 16;
  int lane = threadIdx.x;
  int fr = lane & 15, fk = (lane >> 4) << 4;
  size_t sRow = outgoing ? (size_t)N * C : (size_t)C;
  size_t sK   = outgoing ? (size_t)C : (size_t)N * C;
  v8f acc = {};
  for (int kb = 0; kb < (N >> 5); ++kb) {
    v16h af, bf;
#pragma unroll
    for (int i = 0; i < 16; ++i) {
      size_t kk = (size_t)(kb * 32 + fk + i);
      af[i] = (_Float16)a[(size_t)(i0 + fr) * sRow + kk * sK + c];
      bf[i] = (_Float16)b[(size_t)(j0 + fr) * sRow + kk * sK + c];
    }
    acc = wmma16(af, bf, acc);
  }
#pragma unroll
  for (int r = 0; r < 8; ++r) {
    int gi = i0 + r + ((lane >> 4) << 3);
    int gj = j0 + fr;
    x[((size_t)gi * N + gj) * C + c] = acc[r];
  }
}

// ---------------- Fused outer-product-mean -----------------------------------
// Per block: fixed i, 16 j's.  Stage A: P[j][c*32+d] = sum_s b[s,j,d]*a[s,i,c]
// (WMMA over S).  Stage B: z[i,j,:] += (P @ wo + bo) / norm[i,j].
__global__ __launch_bounds__(256) void opm_kernel(const float* __restrict__ a,
                                                  const float* __restrict__ b,
                                                  const float* __restrict__ wo,
                                                  const float* __restrict__ bo,
                                                  const float* __restrict__ mask,
                                                  float* __restrict__ z,
                                                  int S, int N) {
  const int CO = 32, CZ = 128;
  __shared__ _Float16 P[16][1040];   // row stride 2080B (16B aligned)
  __shared__ float norms[16];
  int j0 = blockIdx.x * 16, i = blockIdx.y;
  int t = threadIdx.x;
  int lane = t & 31, wv = t >> 5;
  int fr = lane & 15, fk = (lane >> 4) << 4;
  if (t < 16) {
    float s = 0.f;
    for (int ss = 0; ss < S; ++ss)
      s += mask[(size_t)ss * N + i] * mask[(size_t)ss * N + j0 + t];
    norms[t] = s + 1e-3f;
  }
  __syncthreads();
  // Stage A: M' = (j,d) = 512 rows, N' = c = 32 cols, K = S.  64 tiles / 8 waves.
  for (int it = 0; it < 8; ++it) {
    int tile = wv + it * 8;
    int mt = tile >> 1, nt = tile & 1;
    v8f acc = {};
    for (int ks = 0; ks < (S >> 5); ++ks) {
      v16h af, bf;
      int m = mt * 16 + fr;
      int jj = m >> 5, d = m & 31;
      int cc = nt * 16 + fr;
#pragma unroll
      for (int u = 0; u < 16; ++u) {
        int s = ks * 32 + fk + u;
        af[u] = (_Float16)b[((size_t)s * N + (j0 + jj)) * CO + d];
        bf[u] = (_Float16)a[((size_t)s * N + i) * CO + cc];
      }
      acc = wmma16(af, bf, acc);
    }
#pragma unroll
    for (int r = 0; r < 8; ++r) {
      int m = mt * 16 + r + ((lane >> 4) << 3);
      int jj = m >> 5, d = m & 31;
      int cc = nt * 16 + fr;
      P[jj][cc * CO + d] = (_Float16)acc[r];
    }
  }
  __syncthreads();
  // Stage B: [16,1024] @ wo[1024,128]; wave wv owns 16 output channels.
  v8f acc = {};
  int col = wv * 16 + fr;
  for (int ks = 0; ks < ((CO * CO) >> 5); ++ks) {
    v16h pf = ld16h(&P[fr][ks * 32 + fk]);
    v16h wf;
#pragma unroll
    for (int u = 0; u < 16; ++u) {
      int kk = ks * 32 + fk + u;
      wf[u] = (_Float16)wo[(size_t)kk * CZ + col];
    }
    acc = wmma16(pf, wf, acc);
  }
#pragma unroll
  for (int r = 0; r < 8; ++r) {
    int jj = r + ((lane >> 4) << 3);
    float val = (acc[r] + bo[col]) / norms[jj];
    z[((size_t)i * N + (j0 + jj)) * CZ + col] += val;
  }
}

// ---------------- Small helpers ----------------------------------------------
__global__ void transpose_kernel(const float* __restrict__ in, float* __restrict__ out,
                                 int A, int B, int C, int accum) {
  int a = blockIdx.x, b = blockIdx.y;
  const float* src = in + ((size_t)a * B + b) * C;
  float* dst = out + ((size_t)b * A + a) * C;
  for (int c = threadIdx.x; c < C; c += blockDim.x) {
    if (accum) dst[c] += src[c]; else dst[c] = src[c];
  }
}

__global__ void maskbias_kernel(const float* __restrict__ mask, float* __restrict__ out,
                                int B, int K, int transposed) {
  int idx = blockIdx.x * blockDim.x + threadIdx.x;
  if (idx >= B * K) return;
  int i = idx / K, j = idx % K;
  float mv = transposed ? mask[(size_t)j * B + i] : mask[idx];
  out[idx] = (mv - 1.0f) * 1e9f;
}

__global__ void gatecombine_kernel(const float* __restrict__ x, const float* __restrict__ g,
                                   const float* __restrict__ rowmask,
                                   float* __restrict__ out, size_t total, int ncol,
                                   int accum) {
  size_t idx = (size_t)blockIdx.x * blockDim.x + threadIdx.x;
  if (idx >= total) return;
  float val = x[idx] * sigmoidf_(g[idx]);
  if (rowmask) val *= rowmask[idx / ncol];
  if (accum) out[idx] += val; else out[idx] = val;
}

// =============================================================================
// Host sequencing.  Input order assumes JAX pytree flattening (sorted keys):
//  0 m | 1 msa_mask | 2 pair_mask | 3.. params | 96 z
//  col 3:bg 4:bo 5:ln_b 6:ln_w 7:wg 8:wk 9:wo 10:wq 11:wv
//  msa_trans 12:b1 13:b2 14:ln_b 15:ln_w 16:w1 17:w2
//  opm 18:ba 19:bb 20:bo 21:ln_b 22:ln_w 23:wa 24:wb 25:wo
//  pair_trans 26:b1 27:b2 28:ln_b 29:ln_w 30:w1 31:w2
//  row 32:bg 33:bo 34:ln_b 35:ln_w 36:ln_z_b 37:ln_z_w 38:w_pair 39:wg 40:wk 41:wo 42:wq 43:wv
//  tri_att_end 44:bg 45:bo 46:ln_b 47:ln_w 48:w_tri 49:wg 50:wk 51:wo 52:wq 53:wv
//  tri_att_start 54..63 (same layout)
//  tri_in 64..79 / tri_out 80..95:
//    +0 ba_g +1 ba_p +2 bb_g +3 bb_p +4 bg +5 bo +6 ln_in_b +7 ln_in_w
//    +8 ln_out_b +9 ln_out_w +10 wa_g +11 wa_p +12 wb_g +13 wb_p +14 wg +15 wo
// =============================================================================
extern "C" void kernel_launch(void* const* d_in, const int* in_sizes, int n_in,
                              void* d_out, int out_size, void* d_ws, size_t ws_size,
                              hipStream_t stream) {
  (void)in_sizes; (void)n_in; (void)out_size; (void)ws_size;
  const int S = 128, N = 256, CM = 256, CZ = 128, HM = 8, HP = 4;
  const float QSCALE = 0.17677669529663687f;  // 1/sqrt(32)
  auto in = [&](int i) { return (const float*)d_in[i]; };

  const float* m_in  = in(0);
  const float* mmask = in(1);
  const float* pmask = in(2);
  const float* z_in  = in(96);

  float* m = (float*)d_out;                 // [S,N,CM]
  float* z = m + (size_t)S * N * CM;        // [N,N,CZ]

  const size_t B8 = 8388608;                // 8M floats = 32MB
  float* ws = (float*)d_ws;
  float* bA  = ws;                          // LN scratch
  float* bB  = ws + 1 * B8;                 // q / a
  float* bC  = ws + 2 * B8;                 // k / b
  float* bD  = ws + 3 * B8;                 // v / x
  float* bE  = ws + 4 * B8;                 // gate lin
  float* bF  = ws + 5 * B8;                 // attn out
  float* bH  = ws + 6 * B8;                 // transposed proj out
  float* bG  = ws + 7 * B8;                 // 33,554,432 floats (transition hidden)
  float* bPB = bG + 33554432;               // pair/tri bias [N*N*H] (<= 524288)
  float* bB1 = bPB + 524288;                // mask bias (<= 65536)
  // total ws needed: 92,864,512 floats = ~355 MB

  hipMemcpyAsync(m, m_in, (size_t)S * N * CM * sizeof(float), hipMemcpyDeviceToDevice, stream);
  hipMemcpyAsync(z, z_in, (size_t)N * N * CZ * sizeof(float), hipMemcpyDeviceToDevice, stream);

  auto LN = [&](const float* x, const float* w, const float* b, float* o, int R, int C) {
    ln_kernel<<<(R + 7) / 8, 256, 0, stream>>>(x, w, b, o, R, C);
  };
  auto GEMM = [&](const float* Ax, const float* Wx, const float* bias, const float* rmask,
                  const float* res, float* o, int M, int Nn, int K,
                  float scale = 1.f, int relu = 0) {
    dim3 g((Nn + 63) / 64, (M + 63) / 64);
    gemm_kernel<<<g, 256, 0, stream>>>(Ax, Wx, bias, rmask, res, o, M, Nn, K, scale, relu);
  };
  auto ATTN = [&](const float* q, const float* k, const float* v, const float* b1,
                  const float* b2, float* o, int Bd, int Q, int K, int H) {
    attn_kernel<<<dim3(Q / 16, H, Bd), 32, 0, stream>>>(q, k, v, b1, b2, o, Bd, Q, K, H);
  };
  auto GATE = [&](const float* x, const float* g, const float* rm, float* o,
                  size_t tot, int ncol, int accum) {
    gatecombine_kernel<<<(unsigned)((tot + 255) / 256), 256, 0, stream>>>(x, g, rm, o, tot, ncol, accum);
  };

  // ---------- MSA row attention with pair bias ----------
  LN(z, in(37), in(36), bA, N * N, CZ);                                  // ln_z(z)
  GEMM(bA, in(38), nullptr, nullptr, nullptr, bPB, N * N, HM, CZ);       // pair bias
  maskbias_kernel<<<(S * N + 255) / 256, 256, 0, stream>>>(mmask, bB1, S, N, 0);
  LN(m, in(35), in(34), bA, S * N, CM);
  GEMM(bA, in(42), nullptr, nullptr, nullptr, bB, S * N, CM, CM, QSCALE); // q
  GEMM(bA, in(40), nullptr, nullptr, nullptr, bC, S * N, CM, CM);         // k
  GEMM(bA, in(43), nullptr, nullptr, nullptr, bD, S * N, CM, CM);         // v
  GEMM(bA, in(39), in(32), nullptr, nullptr, bE, S * N, CM, CM);          // g + bg
  ATTN(bB, bC, bD, bB1, bPB, bF, S, N, N, HM);
  GATE(bF, bE, nullptr, bF, (size_t)S * N * CM, CM, 0);
  GEMM(bF, in(41), in(33), nullptr, m, m, S * N, CM, CM);                 // out + residual

  // ---------- MSA column attention ----------
  transpose_kernel<<<dim3(S, N), 128, 0, stream>>>(m, bG, S, N, CM, 0);
  LN(bG, in(6), in(5), bA, N * S, CM);
  maskbias_kernel<<<(N * S + 255) / 256, 256, 0, stream>>>(mmask, bB1, N, S, 1);
  GEMM(bA, in(10), nullptr, nullptr, nullptr, bB, N * S, CM, CM, QSCALE);
  GEMM(bA, in(8),  nullptr, nullptr, nullptr, bC, N * S, CM, CM);
  GEMM(bA, in(11), nullptr, nullptr, nullptr, bD, N * S, CM, CM);
  GEMM(bA, in(7),  in(3),   nullptr, nullptr, bE, N * S, CM, CM);
  ATTN(bB, bC, bD, bB1, nullptr, bF, N, S, S, HM);
  GATE(bF, bE, nullptr, bF, (size_t)N * S * CM, CM, 0);
  GEMM(bF, in(9), in(4), nullptr, nullptr, bH, N * S, CM, CM);
  transpose_kernel<<<dim3(N, S), 128, 0, stream>>>(bH, m, N, S, CM, 1);   // m += T

  // ---------- MSA transition ----------
  LN(m, in(15), in(14), bA, S * N, CM);
  GEMM(bA, in(16), in(12), nullptr, nullptr, bG, S * N, 4 * CM, CM, 1.f, 1);
  GEMM(bG, in(17), in(13), mmask, m, m, S * N, CM, 4 * CM);

  // ---------- Outer product mean ----------
  LN(m, in(22), in(21), bA, S * N, CM);
  GEMM(bA, in(23), in(18), mmask, nullptr, bB, S * N, 32, CM);            // a
  GEMM(bA, in(24), in(19), mmask, nullptr, bC, S * N, 32, CM);            // b
  opm_kernel<<<dim3(N / 16, N), 256, 0, stream>>>(bB, bC, in(25), in(20), mmask, z, S, N);

  // ---------- Triangle multiplicative updates ----------
  auto TRIMUL = [&](int p, int outgoing) {
    LN(z, in(p + 7), in(p + 6), bA, N * N, CZ);
    GEMM(bA, in(p + 11), in(p + 1), nullptr, nullptr, bB, N * N, CZ, CZ); // a_p
    GEMM(bA, in(p + 10), in(p + 0), nullptr, nullptr, bC, N * N, CZ, CZ); // a_g
    GATE(bB, bC, pmask, bB, (size_t)N * N * CZ, CZ, 0);                   // a
    GEMM(bA, in(p + 13), in(p + 3), nullptr, nullptr, bC, N * N, CZ, CZ); // b_p
    GEMM(bA, in(p + 12), in(p + 2), nullptr, nullptr, bD, N * N, CZ, CZ); // b_g
    GATE(bC, bD, pmask, bC, (size_t)N * N * CZ, CZ, 0);                   // b
    trimul_kernel<<<dim3(N / 16, N / 16, CZ), 32, 0, stream>>>(bB, bC, bD, N, CZ, outgoing);
    LN(bD, in(p + 9), in(p + 8), bE, N * N, CZ);
    GEMM(bE, in(p + 15), in(p + 5), nullptr, nullptr, bF, N * N, CZ, CZ);
    GEMM(bA, in(p + 14), in(p + 4), nullptr, nullptr, bD, N * N, CZ, CZ); // g
    GATE(bF, bD, nullptr, z, (size_t)N * N * CZ, CZ, 1);                  // z += g*proj
  };
  TRIMUL(80, 1);   // tri_out (outgoing)
  TRIMUL(64, 0);   // tri_in  (incoming)

  // ---------- Triangle attention, starting node ----------
  LN(z, in(57), in(56), bA, N * N, CZ);
  GEMM(bA, in(58), nullptr, nullptr, nullptr, bPB, N * N, HP, CZ);        // tri bias
  maskbias_kernel<<<(N * N + 255) / 256, 256, 0, stream>>>(pmask, bB1, N, N, 0);
  GEMM(bA, in(62), nullptr, nullptr, nullptr, bB, N * N, CZ, CZ, QSCALE);
  GEMM(bA, in(60), nullptr, nullptr, nullptr, bC, N * N, CZ, CZ);
  GEMM(bA, in(63), nullptr, nullptr, nullptr, bD, N * N, CZ, CZ);
  GEMM(bA, in(59), in(54), nullptr, nullptr, bE, N * N, CZ, CZ);
  ATTN(bB, bC, bD, bB1, bPB, bF, N, N, N, HP);
  GATE(bF, bE, nullptr, bF, (size_t)N * N * CZ, CZ, 0);
  GEMM(bF, in(61), in(55), nullptr, z, z, N * N, CZ, CZ);

  // ---------- Triangle attention, ending node ----------
  transpose_kernel<<<dim3(N, N), 128, 0, stream>>>(z, bG, N, N, CZ, 0);
  LN(bG, in(47), in(46), bA, N * N, CZ);
  GEMM(bA, in(48), nullptr, nullptr, nullptr, bPB, N * N, HP, CZ);
  maskbias_kernel<<<(N * N + 255) / 256, 256, 0, stream>>>(pmask, bB1, N, N, 1);
  GEMM(bA, in(52), nullptr, nullptr, nullptr, bB, N * N, CZ, CZ, QSCALE);
  GEMM(bA, in(50), nullptr, nullptr, nullptr, bC, N * N, CZ, CZ);
  GEMM(bA, in(53), nullptr, nullptr, nullptr, bD, N * N, CZ, CZ);
  GEMM(bA, in(49), in(44), nullptr, nullptr, bE, N * N, CZ, CZ);
  ATTN(bB, bC, bD, bB1, bPB, bF, N, N, N, HP);
  GATE(bF, bE, nullptr, bF, (size_t)N * N * CZ, CZ, 0);
  GEMM(bF, in(51), in(45), nullptr, nullptr, bH, N * N, CZ, CZ);
  transpose_kernel<<<dim3(N, N), 128, 0, stream>>>(bH, z, N, N, CZ, 1);   // z += T

  // ---------- Pair transition ----------
  LN(z, in(29), in(28), bA, N * N, CZ);
  GEMM(bA, in(30), in(26), nullptr, nullptr, bG, N * N, 4 * CZ, CZ, 1.f, 1);
  GEMM(bG, in(31), in(27), pmask, z, z, N * N, CZ, 4 * CZ);
}
